// ImageEncoderViT_72756745994845
// MI455X (gfx1250) — matrix operand order repaired and checked by reading, not compiled, not run on hardware
//
#include <hip/hip_runtime.h>
#include <hip/hip_bf16.h>
#include <math.h>

typedef __attribute__((ext_vector_type(16))) _Float16 v16h;
typedef __attribute__((ext_vector_type(8)))  _Float16 v8h;
typedef __attribute__((ext_vector_type(4)))  _Float16 v4h;
typedef __attribute__((ext_vector_type(8)))  float    v8f;
typedef __attribute__((ext_vector_type(4)))  float    f4;

#define NHEADS 8
#define DHEAD  24
#define CDIM   192
#define HID    768
#define BB     4
#define HH     128
#define WW     128
#define NTOK   (BB*HH*WW)            /* 65536 tokens per stream        */
#define NWIN   (NTOK/4)              /* 16384 windows per stream       */
#define MT     64                    /* token tile per workgroup (MLP) */
#define MB     (MT/16)               /* 4 M-blocks                     */

union V16 { v16h v; v8h h[2]; };

// ---------------------------------------------------------------------------
// Kernel 1: weight transpose + f32->f16.  src[s][R][Cc] -> dst[s][Cc][R]
// ---------------------------------------------------------------------------
__global__ __launch_bounds__(256) void wconv_kernel(const float* __restrict__ src,
                                                    _Float16* __restrict__ dst,
                                                    int R, int Cc) {
    int s = blockIdx.y;
    int total = R * Cc;
    const float* sp = src + (long)s * total;
    _Float16*    dp = dst + (long)s * total;
    for (int i = blockIdx.x * 256 + threadIdx.x; i < total; i += gridDim.x * 256) {
        int r = i / Cc, c = i % Cc;
        dp[(long)c * R + r] = (_Float16)sp[i];
    }
}

// ---------------------------------------------------------------------------
// Kernel 2: windowed cross-attention, one wave32 per window.
// lane = head*4 + token_i.  Streams: q={r,g,b,ir}, kv={g,b,ir,g}.
// Output: f32 token-major [s][t=win*4+ti][192]
// ---------------------------------------------------------------------------
__global__ __launch_bounds__(256) void attn_kernel(const float* __restrict__ r,
                                                   const float* __restrict__ g,
                                                   const float* __restrict__ bch,
                                                   const float* __restrict__ ir,
                                                   float* __restrict__ aF32) {
    int gwid = (blockIdx.x * 256 + threadIdx.x) >> 5;   // window id
    int lane = threadIdx.x & 31;
    int s    = gwid >> 14;
    int wlin = gwid & (NWIN - 1);

    const float* Q  = (s == 0) ? r : (s == 1) ? g : (s == 2) ? bch : ir;
    const float* KV = (s == 0) ? g : (s == 1) ? bch : (s == 2) ? ir : g;

    int head = lane >> 2, ti = lane & 3;
    int co   = head * DHEAD;

    int b   = wlin >> 12;
    int rem = wlin & 4095;
    int wy  = rem >> 6, wx = rem & 63;
    long toff[4];
#pragma unroll
    for (int n = 0; n < 4; ++n) {
        int y = wy * 2 + (n >> 1), x = wx * 2 + (n & 1);
        toff[n] = (((long)b * HH + y) * WW + x) * CDIM + co;
    }

    f4 qv[6];
    const float* qp = Q + toff[ti];
#pragma unroll
    for (int i = 0; i < 6; ++i) qv[i] = *(const f4*)(qp + i * 4);

    float sc[4];
#pragma unroll
    for (int j = 0; j < 4; ++j) {
        const float* kp = KV + toff[j];
        float dot = 0.f;
#pragma unroll
        for (int i = 0; i < 6; ++i) {
            f4 kv = *(const f4*)(kp + i * 4);
            dot += qv[i][0]*kv[0] + qv[i][1]*kv[1] + qv[i][2]*kv[2] + qv[i][3]*kv[3];
        }
        sc[j] = dot * 0.20412414523193154f;   // 1/sqrt(24)
    }
    float mx = fmaxf(fmaxf(sc[0], sc[1]), fmaxf(sc[2], sc[3]));
    float p[4], sum = 0.f;
#pragma unroll
    for (int j = 0; j < 4; ++j) { p[j] = __expf(sc[j] - mx); sum += p[j]; }
    float inv = 1.f / sum;

    f4 acc[6] = {};
#pragma unroll
    for (int j = 0; j < 4; ++j) {
        float pj = p[j] * inv;
        const float* vp = KV + toff[j];
#pragma unroll
        for (int i = 0; i < 6; ++i) {
            f4 vv = *(const f4*)(vp + i * 4);
            acc[i] += pj * vv;
        }
    }

    long orow = ((long)s * NTOK + (long)wlin * 4 + ti) * CDIM + co;
#pragma unroll
    for (int i = 0; i < 6; ++i) *(f4*)(aF32 + orow + i * 4) = acc[i];
}

// ---------------------------------------------------------------------------
// WMMA fragment helpers (16x16x32 f16 -> f32)
// ---------------------------------------------------------------------------
__device__ __forceinline__ v16h frag_a_lds(const _Float16* base, int m0, int rs,
                                           int k0, int lane) {
    int m = m0 + (lane & 15);
    int sel = lane >> 4;                          // lanes 16-31: K+8 / K+24
    const _Float16* p = base + m * rs + k0 + sel * 8;
    V16 t; t.h[0] = *(const v8h*)p; t.h[1] = *(const v8h*)(p + 16);
    return t.v;
}
__device__ __forceinline__ v16h frag_b_glb(const _Float16* baseT, int n0, int rs,
                                           int k0, int lane) {
    int n = n0 + (lane & 15);
    int ksel = lane >> 4;                         // lanes 16-31: K+8 / K+24
    const _Float16* p = baseT + (long)n * rs + k0 + ksel * 8;
    V16 t; t.h[0] = *(const v8h*)p; t.h[1] = *(const v8h*)(p + 16);
    return t.v;
}
__device__ __forceinline__ v8f wmma_f16(v16h a, v16h b, v8f c) {
    return __builtin_amdgcn_wmma_f32_16x16x32_f16(false, a, false, b,
                                                  (short)0, c, false, false);
}

// Branch-free fast GELU (tanh form), v_exp_f32 + v_rcp_f32, saturates at +/-inf
__device__ __forceinline__ float fast_gelu(float x) {
    float u = x * (0.7978845608f + 0.0356774081f * x * x);
    float e = __expf(2.0f * u);                       // exp(2u)
    float t = 1.0f - 2.0f * __builtin_amdgcn_rcpf(e + 1.0f);  // tanh(u)
    return 0.5f * x * (1.0f + t);
}

// ---------------------------------------------------------------------------
// Kernel 3: fused MLP  out = attn + fc2(gelu(fc1(attn)))
// workgroup: 256 thr (8 waves), MT=64 tokens.  LDS: A 24KB + H 96KB = 120KB
// Waves own N-blocks and sweep all M-blocks so B fragments are register-reused.
// ---------------------------------------------------------------------------
__global__ __launch_bounds__(256) void mlp_kernel(const float* __restrict__ aF32,
                                                  const _Float16* __restrict__ w1t,
                                                  const _Float16* __restrict__ w2t,
                                                  const float* __restrict__ b1,
                                                  const float* __restrict__ b2,
                                                  float* __restrict__ out) {
    __shared__ _Float16 sA[MT * CDIM];   // 24 KB
    __shared__ _Float16 sH[MT * HID];    // 96 KB

    int s        = blockIdx.y;
    long tokBase = (long)blockIdx.x * MT;       // stream-local token base
    int tid = threadIdx.x, lane = tid & 31, wave = tid >> 5;

    // ---- stage A: f32 attention tile -> f16 in LDS --------------------
    const float* aSrc = aF32 + ((long)s * NTOK + tokBase) * CDIM;
    for (int i = tid; i < MT * CDIM / 4; i += 256) {
        f4 v = *(const f4*)(aSrc + i * 4);
        v4h h; h[0] = (_Float16)v[0]; h[1] = (_Float16)v[1];
        h[2] = (_Float16)v[2]; h[3] = (_Float16)v[3];
        *(v4h*)(sA + i * 4) = h;
    }
    __syncthreads();

    // ---- fc1: h = gelu(A @ W1 + b1), h in LDS (f16) -------------------
    // wave owns nb, sweeps 4 M-blocks: B frag loaded once, reused 4x
    const _Float16* W1 = w1t + (long)s * HID * CDIM;   // [768][192]
    for (int nb = wave; nb < HID / 16; nb += 8) {
        v8f acc[MB] = {};
#pragma unroll
        for (int kk = 0; kk < CDIM / 32; ++kk) {
            int k0 = kk * 32;
            v16h bfr = frag_b_glb(W1, nb * 16, CDIM, k0, lane);
            __builtin_prefetch(W1 + (long)(nb * 16 + (lane & 15)) * CDIM + k0 + 32, 0, 1);
#pragma unroll
            for (int mb = 0; mb < MB; ++mb) {
                v16h a = frag_a_lds(sA, mb * 16, CDIM, k0, lane);
                acc[mb] = wmma_f16(a, bfr, acc[mb]);
            }
        }
        int n = nb * 16 + (lane & 15);
        float bias = b1[s * HID + n];
#pragma unroll
        for (int mb = 0; mb < MB; ++mb) {
#pragma unroll
            for (int j = 0; j < 8; ++j) {
                int m = mb * 16 + j + (lane >> 4) * 8;
                sH[m * HID + n] = (_Float16)fast_gelu(acc[mb][j] + bias);
            }
        }
    }
    __syncthreads();

    // ---- fc2: out = h @ W2 + b2 + residual, un-window store -----------
    // task = (nb, m-half): 12 nb x 2 halves = 24 tasks, 3 per wave, B reuse 2x
    const _Float16* W2 = w2t + (long)s * CDIM * HID;   // [192][768]
    const float* res = aSrc;
    for (int t = wave; t < (CDIM / 16) * 2; t += 8) {
        int nb = t >> 1, mh = t & 1;
        v8f acc[2] = {};
#pragma unroll 4
        for (int kk = 0; kk < HID / 32; ++kk) {
            int k0 = kk * 32;
            v16h bfr = frag_b_glb(W2, nb * 16, HID, k0, lane);
            __builtin_prefetch(W2 + (long)(nb * 16 + (lane & 15)) * HID + k0 + 32, 0, 1);
#pragma unroll
            for (int mm = 0; mm < 2; ++mm) {
                v16h a = frag_a_lds(sH, (mh * 2 + mm) * 16, HID, k0, lane);
                acc[mm] = wmma_f16(a, bfr, acc[mm]);
            }
        }
        int n = nb * 16 + (lane & 15);
        float bias = b2[s * CDIM + n];
#pragma unroll
        for (int mm = 0; mm < 2; ++mm) {
#pragma unroll
            for (int j = 0; j < 8; ++j) {
                int m = (mh * 2 + mm) * 16 + j + (lane >> 4) * 8;
                long tok = tokBase + m;
                float val = acc[mm][j] + bias + res[(long)m * CDIM + n];
                int w  = (int)(tok >> 2), nt = (int)(tok & 3);
                int bb = w >> 12;
                int rem = w & 4095;
                int wy = rem >> 6, wx = rem & 63;
                int y = wy * 2 + (nt >> 1), x = wx * 2 + (nt & 1);
                out[((((long)s * BB + bb) * HH + y) * WW + x) * CDIM + n] = val;
            }
        }
    }
}

// ---------------------------------------------------------------------------
extern "C" void kernel_launch(void* const* d_in, const int* in_sizes, int n_in,
                              void* d_out, int out_size, void* d_ws, size_t ws_size,
                              hipStream_t stream) {
    const float* r    = (const float*)d_in[0];
    const float* g    = (const float*)d_in[1];
    const float* b    = (const float*)d_in[2];
    const float* ir   = (const float*)d_in[3];
    const float* fc1w = (const float*)d_in[4];
    const float* fc1b = (const float*)d_in[5];
    const float* fc2w = (const float*)d_in[6];
    const float* fc2b = (const float*)d_in[7];

    char* ws = (char*)d_ws;
    const size_t AF32_BYTES = (size_t)4 * NTOK * CDIM * sizeof(float);   // 201.3 MB
    const size_t WT_BYTES   = (size_t)4 * HID * CDIM * sizeof(_Float16); // 1.18 MB
    float*    aF32 = (float*)ws;
    _Float16* w1t  = (_Float16*)(ws + AF32_BYTES);
    _Float16* w2t  = (_Float16*)(ws + AF32_BYTES + WT_BYTES);

    // weights: f32 -> f16 transposed (so B-fragments are contiguous)
    wconv_kernel<<<dim3(288, 4), 256, 0, stream>>>(fc1w, w1t, CDIM, HID);
    wconv_kernel<<<dim3(288, 4), 256, 0, stream>>>(fc2w, w2t, HID, CDIM);

    // attention: one wave per window, 4 streams * 16384 windows
    attn_kernel<<<(4 * NWIN) / 8, 256, 0, stream>>>(r, g, b, ir, aF32);

    // fused MLP + residual + un-windowing
    mlp_kernel<<<dim3(NTOK / MT, 4), 256, 0, stream>>>(aF32, w1t, w2t,
                                                       fc1b, fc2b, (float*)d_out);
}
